// GlimpseNetwork_42417097016879
// MI455X (gfx1250) — matrix operand order repaired and checked by reading, not compiled, not run on hardware
//
#include <hip/hip_runtime.h>
#include <hip/hip_bf16.h>

typedef __attribute__((ext_vector_type(16))) _Float16 v16h;
typedef __attribute__((ext_vector_type(4)))  _Float16 v4h;
typedef __attribute__((ext_vector_type(8)))  float    v8f;
typedef __attribute__((ext_vector_type(4)))  float    v4f;

union AFrag { v16h h; v4f f[2]; };

#define BB   128
#define DIN  36864
#define HG   128
#define NOUT 256
#define KSPLIT 8

// ---- prep: W1 (DIN x 128, f32) -> W1t (128 x DIN, f16, K-major) ---------------
// LDS-tiled 32x32 transpose: coalesced reads (128B/wave) and writes (64B/wave).
__global__ __launch_bounds__(256) void k_prep_w1t(const float* __restrict__ W1,
                                                  _Float16* __restrict__ W1t) {
  __shared__ _Float16 tile[32 * 33];               // +1 pad vs bank conflicts
  int dt = blockIdx.x >> 2;                        // 0..1151  (DIN/32 tiles)
  int nt = blockIdx.x & 3;                         // 0..3     (128/32 tiles)
  int d0 = dt * 32, n0 = nt * 32;
  int tx = threadIdx.x & 31;                       // column within tile
  int ty = threadIdx.x >> 5;                       // 0..7
#pragma unroll
  for (int r = 0; r < 4; ++r) {
    int dl = ty + r * 8;                           // 0..31
    tile[dl * 33 + tx] = (_Float16)W1[(size_t)(d0 + dl) * HG + n0 + tx];
  }
  __syncthreads();
#pragma unroll
  for (int r = 0; r < 4; ++r) {
    int nl = ty + r * 8;
    W1t[(size_t)(n0 + nl) * DIN + d0 + tx] = tile[tx * 33 + nl];
  }
}

// ---- foveate: gather patches, avg-pool, emit phi (B x DIN, f16) ---------------
// 4 adjacent j-outputs per thread: contiguous row reads, one aligned v4h store.
__global__ __launch_bounds__(256) void k_foveate(const float* __restrict__ x,
                                                 const float* __restrict__ l,
                                                 _Float16* __restrict__ phi) {
  const int NQ = DIN / 4;                          // 9216 quads per batch image
  int tid = blockIdx.x * blockDim.x + threadIdx.x;
  if (tid >= BB * NQ) return;
  int b  = tid / NQ;
  int q  = tid - b * NQ;
  int jq = q & 15;                                 // j / 4
  int i  = (q >> 4) & 63;
  int ch = q >> 10;                                // 0..8  (k*3 + c)
  int k = ch / 3, c = ch - 3 * k;
  int f = 1 << k;                                  // pool factor 1,2,4
  int size = 64 << k;                              // patch size 64,128,256
  float lx = l[2 * b], ly = l[2 * b + 1];
  int stx = (int)(0.5f * ((lx + 1.0f) * 512.0f));  // trunc, matches astype(int32)
  int sty = (int)(0.5f * ((ly + 1.0f) * 512.0f));
  int br = sty - (size >> 1) + i * f;
  int cb = stx - (size >> 1) + (jq * 4) * f;
  const float* img = x + (size_t)(b * 3 + c) * 512 * 512;
  float s0 = 0.0f, s1 = 0.0f, s2 = 0.0f, s3 = 0.0f;
  for (int fi = 0; fi < f; ++fi) {
    int gr = br + fi;
    if (gr < 0 || gr >= 512) continue;             // zero padding contributes 0
    const float* row = img + (size_t)gr * 512;
    for (int fj = 0; fj < f; ++fj) {
      int g0 = cb + fj;
      int g1 = g0 + f, g2 = g1 + f, g3 = g2 + f;
      if (g0 >= 0 && g0 < 512) s0 += row[g0];
      if (g1 >= 0 && g1 < 512) s1 += row[g1];
      if (g2 >= 0 && g2 < 512) s2 += row[g2];
      if (g3 >= 0 && g3 < 512) s3 += row[g3];
    }
  }
  float inv = 1.0f / (float)(f * f);
  v4h o;
  o.x = (_Float16)(s0 * inv);
  o.y = (_Float16)(s1 * inv);
  o.z = (_Float16)(s2 * inv);
  o.w = (_Float16)(s3 * inv);
  *(v4h*)(phi + (size_t)b * DIN + (size_t)ch * 4096 + i * 64 + jq * 4) = o;
}

// ---- small prep: W34t (256x256 f16, K-major, [W3;W4]), l_out -> Acat[:,128:], b34
__global__ __launch_bounds__(256) void k_small_prep(const float* __restrict__ l,
                                                    const float* __restrict__ W2,
                                                    const float* __restrict__ b2,
                                                    const float* __restrict__ W3,
                                                    const float* __restrict__ b3,
                                                    const float* __restrict__ W4,
                                                    const float* __restrict__ b4,
                                                    _Float16* __restrict__ Acat,
                                                    _Float16* __restrict__ W34t,
                                                    float* __restrict__ b34) {
  int idx = blockIdx.x * blockDim.x + threadIdx.x;
  if (idx < 256 * 256) {                 // W34t[n][k], k = stacked H_G|H_L axis
    int n = idx >> 8, k = idx & 255;
    float v = (k < 128) ? W3[k * 256 + n] : W4[(k - 128) * 256 + n];
    W34t[(size_t)n * 256 + k] = (_Float16)v;
  }
  if (idx < 128 * 128) {                 // l_out = relu(l @ W2 + b2), K=2
    int b = idx >> 7, h = idx & 127;
    float v = l[b * 2] * W2[h] + l[b * 2 + 1] * W2[128 + h] + b2[h];
    v = v > 0.0f ? v : 0.0f;
    Acat[(size_t)b * NOUT + 128 + h] = (_Float16)v;
  }
  if (idx < 256) b34[idx] = b3[idx] + b4[idx];
}

// ---- WMMA tile accumulator (f16 in, f32 acc) ----------------------------------
// A: M-major rows of ld halves, B: N-major rows of ld halves (K-major both).
// Fragment k order is contiguous, k = 16*(lane/16) + s, same for A and B (wave32, 16x16x32 f16).
__device__ inline v8f wmma_acc(const _Float16* __restrict__ A,
                               const _Float16* __restrict__ B,
                               int ld, int kStart, int kCount,
                               int mBase, int nBase) {
  int lane   = threadIdx.x & 31;
  int laneLo = lane & 15;
  int laneHi = lane >> 4;
  const _Float16* arow = A + (size_t)(mBase + laneLo) * ld + kStart + laneHi * 16;
  const _Float16* brow = B + (size_t)(nBase + laneLo) * ld + kStart + laneHi * 16;
  v8f acc = {};
  for (int k0 = 0; k0 < kCount; k0 += 32) {
    AFrag fa;                                     // A: K {0..15} / {16..31}
    const v4f* pa = (const v4f*)(arow + k0);
    fa.f[0] = pa[0];                              // 8 halves (16B)
    fa.f[1] = pa[1];                              // +8 halves (16B)
    v16h fb = *(const v16h*)(brow + k0);          // B: contiguous 16 halves (32B)
    acc = __builtin_amdgcn_wmma_f32_16x16x32_f16(
        /*neg_a=*/false, fa.h, /*neg_b=*/false, fb,
        /*c_mod=*/(short)0, acc, /*reuse_a=*/false, /*reuse_b=*/false);
  }
  return acc;
}

// ---- GEMM1 split-K: phi(128xDIN) x W1t -> partials ----------------------------
__global__ __launch_bounds__(256) void k_gemm1(const _Float16* __restrict__ phi,
                                               const _Float16* __restrict__ W1t,
                                               float* __restrict__ partial) {
  int wid = blockIdx.x * 8 + (threadIdx.x >> 5);  // 0..511
  int s = wid >> 6;                               // K segment 0..7
  int t = wid & 63;                               // output tile 0..63
  int mBase = (t >> 3) * 16, nBase = (t & 7) * 16;
  const int kLen = DIN / KSPLIT;                  // 4608
  v8f acc = wmma_acc(phi, W1t, DIN, s * kLen, kLen, mBase, nBase);
  int lane = threadIdx.x & 31;
  int laneLo = lane & 15, laneHi = lane >> 4;
  int n = nBase + laneLo;
  float* pbase = partial + (size_t)s * (HG * HG);
#pragma unroll
  for (int r = 0; r < 8; ++r) {
    int m = mBase + laneHi * 8 + r;
    pbase[(size_t)m * HG + n] = acc[r];
  }
}

// ---- reduce split-K + bias + relu -> Acat[:, :128] (f16) ----------------------
__global__ __launch_bounds__(256) void k_gemm1_post(const float* __restrict__ partial,
                                                    const float* __restrict__ b1,
                                                    _Float16* __restrict__ Acat) {
  int idx = blockIdx.x * blockDim.x + threadIdx.x;  // 0..16383
  if (idx >= HG * HG) return;
  int m = idx >> 7, n = idx & 127;
  float v = b1[n];
#pragma unroll
  for (int s = 0; s < KSPLIT; ++s) v += partial[(size_t)s * (HG * HG) + idx];
  v = v > 0.0f ? v : 0.0f;
  Acat[(size_t)m * NOUT + n] = (_Float16)v;
}

// ---- GEMM2: Acat(128x256) x [W3;W4]t + b34, relu -> out (f32) -----------------
__global__ __launch_bounds__(256) void k_gemm2(const _Float16* __restrict__ Acat,
                                               const _Float16* __restrict__ W34t,
                                               const float* __restrict__ b34,
                                               float* __restrict__ out) {
  int wid = blockIdx.x * 8 + (threadIdx.x >> 5);  // 0..127
  int mBase = (wid >> 4) * 16;                    // 8 M tiles
  int nBase = (wid & 15) * 16;                    // 16 N tiles
  v8f acc = wmma_acc(Acat, W34t, NOUT, 0, NOUT, mBase, nBase);
  int lane = threadIdx.x & 31;
  int laneLo = lane & 15, laneHi = lane >> 4;
  int n = nBase + laneLo;
  float bias = b34[n];
#pragma unroll
  for (int r = 0; r < 8; ++r) {
    int m = mBase + laneHi * 8 + r;
    float v = acc[r] + bias;
    v = v > 0.0f ? v : 0.0f;
    out[(size_t)m * NOUT + n] = v;
  }
}

extern "C" void kernel_launch(void* const* d_in, const int* in_sizes, int n_in,
                              void* d_out, int out_size, void* d_ws, size_t ws_size,
                              hipStream_t stream) {
  (void)in_sizes; (void)n_in; (void)out_size; (void)ws_size;
  const float* x  = (const float*)d_in[0];
  const float* l  = (const float*)d_in[1];
  const float* W1 = (const float*)d_in[2];
  const float* b1 = (const float*)d_in[3];
  const float* W2 = (const float*)d_in[4];
  const float* b2 = (const float*)d_in[5];
  const float* W3 = (const float*)d_in[6];
  const float* b3 = (const float*)d_in[7];
  const float* W4 = (const float*)d_in[8];
  const float* b4 = (const float*)d_in[9];
  float* out = (float*)d_out;

  // workspace layout (all offsets 32B-aligned)
  char* ws = (char*)d_ws;
  _Float16* W1t     = (_Float16*)(ws);                 //  9,437,184 B
  _Float16* phi     = (_Float16*)(ws + 9437184);       //  9,437,184 B
  _Float16* Acat    = (_Float16*)(ws + 18874368);      //     65,536 B (128x256)
  _Float16* W34t    = (_Float16*)(ws + 18939904);      //    131,072 B (256x256)
  float*    b34     = (float*)   (ws + 19070976);      //      1,024 B
  float*    partial = (float*)   (ws + 19072000);      //    524,288 B (8x128x128)

  k_prep_w1t<<<(DIN / 32) * (HG / 32), 256, 0, stream>>>(W1, W1t);
  k_foveate <<<(BB * (DIN / 4) + 255) / 256, 256, 0, stream>>>(x, l, phi);
  k_small_prep<<<256, 256, 0, stream>>>(l, W2, b2, W3, b3, W4, b4, Acat, W34t, b34);
  k_gemm1<<<64, 256, 0, stream>>>(phi, W1t, partial);
  k_gemm1_post<<<(HG * HG + 255) / 256, 256, 0, stream>>>(partial, b1, Acat);
  k_gemm2<<<16, 256, 0, stream>>>(Acat, W34t, b34, out);
}
